// NeuronMixtralDecoderLayer_4492535791707
// MI455X (gfx1250) — matrix operand order repaired and checked by reading, not compiled, not run on hardware
//
#include <hip/hip_runtime.h>
#include <hip/hip_bf16.h>
#include <math.h>

// ---------------------------------------------------------------------------
// MI455X (gfx1250, wave32) Mixtral decoder layer.
// All GEMMs run on V_WMMA_F32_16X16X32_BF16 (fp32 accumulate). fp32 weights
// are converted to bf16 with packed converts during LDS staging (weights are
// streamed exactly once -> bandwidth-bound, 352MB @ 23.3TB/s ~ 15us floor).
// The (already-bf16) A operand is staged via the Tensor Data Mover with
// double buffering, overlapping the activation stream with WMMA compute.
// ---------------------------------------------------------------------------

typedef __bf16 bf16;
typedef __attribute__((ext_vector_type(16))) __bf16 v16bf;
typedef __attribute__((ext_vector_type(8)))  __bf16 v8bf;
typedef __attribute__((ext_vector_type(2)))  __bf16 v2bf;
typedef __attribute__((ext_vector_type(8)))  float  v8f;

typedef __attribute__((ext_vector_type(4))) unsigned tdm_v4u;
typedef __attribute__((ext_vector_type(4))) int      tdm_v4i;
typedef __attribute__((ext_vector_type(8))) int      tdm_v8i;

#if defined(__gfx1250__) && __has_builtin(__builtin_amdgcn_tensor_load_to_lds) && \
    __has_builtin(__builtin_amdgcn_s_wait_tensorcnt)
#define USE_TDM 1
#else
#define USE_TDM 0
#endif

static constexpr int B_ = 2, S_ = 1024, H_ = 1024;
static constexpr int NH_ = 16, NKV_ = 4, HD_ = 64;
static constexpr int E_ = 8, I_ = 3584;
static constexpr int T_ = B_ * S_;
static constexpr float EPS_ = 1e-5f;
static constexpr float THETA_ = 1000000.0f;

__device__ __forceinline__ bf16 f2bf(float f) { return (bf16)f; }

__device__ __forceinline__ v8f vzero8() {
    v8f v;
    for (int i = 0; i < 8; i++) v[i] = 0.f;
    return v;
}

__device__ __forceinline__ v16bf cat8(v8bf lo, v8bf hi) {
    return __builtin_shufflevector(lo, hi, 0,1,2,3,4,5,6,7,8,9,10,11,12,13,14,15);
}

#if USE_TDM
// Build a D# for a 2D bf16 tile load (tileX fastest-varying, row stride in
// elements) and kick the Tensor Data Mover.  Bitfields per CDNA5 ISA 8.3/8.4.
// This toolchain's builtin is the 6-arg form:
//   (uint32x4 g0, int32x8 g1, int32x4 g2, int32x4 g3, int32x8 extra, i32 cpol)
__device__ __forceinline__ void tdm_load_2d_bf16(const void* gsrc, unsigned lds_off,
                                                 int rowElems, int nRows,
                                                 int tileX, int tileY)
{
    unsigned long long ga = (unsigned long long)(uintptr_t)gsrc;
    tdm_v4u g0;
    g0[0] = 1u;                                              // count=1, user D#
    g0[1] = lds_off;                                         // lds_addr
    g0[2] = (unsigned)ga;                                    // global_addr lo
    g0[3] = (unsigned)((ga >> 32) & 0x01ffffffu) | 0x80000000u; // addr hi | type=2
    tdm_v8i g1;
    g1[0] = (int)(1u << 16);                                 // data_size = 2 bytes
    g1[1] = (int)(((unsigned)rowElems & 0xffffu) << 16);     // tensor_dim0 lo16
    g1[2] = (int)((((unsigned)rowElems >> 16) & 0xffffu) |
                  (((unsigned)nRows & 0xffffu) << 16));      // dim0 hi | dim1 lo
    g1[3] = (int)((((unsigned)nRows >> 16) & 0xffffu) |
                  (((unsigned)tileX & 0xffffu) << 16));      // dim1 hi | tile_dim0
    g1[4] = (int)((unsigned)tileY & 0xffffu);                // tile_dim1 (dim2=0)
    g1[5] = rowElems;                                        // tensor_dim0_stride
    g1[6] = 0;
    g1[7] = 0;
    tdm_v4i gz4 = (tdm_v4i)0;                                // 2D: groups 2/3 unused
    tdm_v8i gz8 = (tdm_v8i)0;
    __builtin_amdgcn_tensor_load_to_lds(g0, g1, gz4, gz4, gz8, 0);
}
#endif

// ---------------------------------------------------------------------------
// Generic tiled GEMM: C[M,N] = A_bf16[M,K] x B_fp32->bf16[K,N], fp32 accum.
// 256 threads = 8 waves; block tile 128x128x32; wave tile 64x32 (4x2 frags).
// A tile double-buffered via TDM; B tile converted with packed bf16 stores.
// ---------------------------------------------------------------------------
#define BM 128
#define BN 128
#define BK 32

enum { MODE_STORE = 0, MODE_RESID = 1, MODE_SILU = 2, MODE_MOE = 3 };

template<int MODE>
__global__ __launch_bounds__(256)
void gemm_bf16_wmma(const bf16* __restrict__ A, const float* __restrict__ Bw,
                    void* __restrict__ Cv, const float* __restrict__ aux,
                    int M, int N, int K, int auxStride, int auxOff)
{
    __shared__ __align__(32) bf16 As[2][BM][BK];    // row-major [m][k], 2 bufs
    __shared__ __align__(32) bf16 Bs[BN][BK];       // transposed [n][k]

    const int tid  = threadIdx.x;
    const int lane = tid & 31;
    const int wave = tid >> 5;
    const int wm   = wave & 1;    // 2 row-groups of 64
    const int wn   = wave >> 1;   // 4 col-groups of 32
    const int m0   = blockIdx.y * BM;
    const int n0   = blockIdx.x * BN;
    const int lh   = lane & 15;   // N/M within 16x16 tile
    const int lg   = lane >> 4;   // lane-half selector

    v8f acc[4][2];
    for (int i = 0; i < 4; i++)
        for (int j = 0; j < 2; j++) acc[i][j] = vzero8();

#if USE_TDM
    const unsigned asOff0 = (unsigned)(uintptr_t)&As[0][0][0];
    const unsigned asOff1 = (unsigned)(uintptr_t)&As[1][0][0];
    if (wave == 0)  // one DMA per block (TDM ignores EXEC)
        tdm_load_2d_bf16(A + (size_t)m0 * K, asOff0, K, M, BK, BM);
#endif

    int buf = 0;
    for (int k0 = 0; k0 < K; k0 += BK) {
#if !USE_TDM
        {   // fallback: stage A tile with 32B vector copies
            int r  = tid >> 1;
            int cb = (tid & 1) * 16;
            *(v16bf*)&As[buf][r][cb] =
                *(const v16bf*)(A + (size_t)(m0 + r) * K + k0 + cb);
        }
#endif
        // Stage B tile fp32 -> bf16, transposed into Bs[n][k].
        // Each thread: 2 consecutive k rows x 8 n -> packed dword stores.
        {
            int kp = (tid & 15) * 2;              // 0,2,...,30
            int nb = (tid >> 4) * 8;              // 0..120
            const float* s0 = Bw + (size_t)(k0 + kp) * N + n0 + nb;
            const float* s1 = s0 + N;
            #pragma unroll
            for (int j = 0; j < 8; j++) {
                v2bf p;
                p[0] = f2bf(s0[j]);
                p[1] = f2bf(s1[j]);
                *(v2bf*)&Bs[nb + j][kp] = p;      // 4B-aligned packed store
            }
            if (k0 + BK < K)                      // emits global_prefetch_b8
                __builtin_prefetch(Bw + (size_t)(k0 + BK + kp) * N + n0 + nb, 0, 1);
        }
#if USE_TDM
        if (wave == 0) {
            if (k0 + BK < K) {
                // Prefetch next A tile into the other buffer, then wait for
                // the current one (TDM ops retire in order: cnt<=1 => done).
                tdm_load_2d_bf16(A + (size_t)m0 * K + (k0 + BK),
                                 buf ? asOff0 : asOff1, K, M, BK, BM);
                __builtin_amdgcn_s_wait_tensorcnt(1);
            } else {
                __builtin_amdgcn_s_wait_tensorcnt(0);
            }
        }
#endif
        __syncthreads();

        // B fragments: lane e-th element is K = lg*16 + e, column = lh.
        v16bf bfrag[2];
        #pragma unroll
        for (int ni = 0; ni < 2; ni++)
            bfrag[ni] = *(const v16bf*)&Bs[wn * 32 + ni * 16 + lh][lg * 16];

        // A fragments: K(e) = lg*16 + e, row = lh.
        #pragma unroll
        for (int mi = 0; mi < 4; mi++) {
            int mm = wm * 64 + mi * 16 + lh;
            v8bf lo = *(const v8bf*)&As[buf][mm][lg * 16];
            v8bf hi = *(const v8bf*)&As[buf][mm][lg * 16 + 8];
            v16bf afrag = cat8(lo, hi);
            #pragma unroll
            for (int ni = 0; ni < 2; ni++)
                acc[mi][ni] = __builtin_amdgcn_wmma_f32_16x16x32_bf16(
                    false, afrag, false, bfrag[ni], (short)0, acc[mi][ni],
                    false, false);
        }
        __syncthreads();
        buf ^= 1;
    }

    // Epilogue: C layout -> N = lh, M = vgpr_i + 8*lg.
    #pragma unroll
    for (int mi = 0; mi < 4; mi++) {
        #pragma unroll
        for (int ni = 0; ni < 2; ni++) {
            int n = n0 + wn * 32 + ni * 16 + lh;
            #pragma unroll
            for (int i = 0; i < 8; i++) {
                int m = m0 + wm * 64 + mi * 16 + lg * 8 + i;
                float v = acc[mi][ni][i];
                size_t idx = (size_t)m * N + n;
                if (MODE == MODE_STORE) {
                    ((float*)Cv)[idx] = v;
                } else if (MODE == MODE_RESID) {
                    ((float*)Cv)[idx] = v + aux[idx];
                } else if (MODE == MODE_SILU) {
                    float g  = aux[idx];
                    float sg = g / (1.0f + __expf(-g));
                    ((bf16*)Cv)[idx] = f2bf(sg * v);
                } else { // MODE_MOE: out += routerWeight[token] * acc
                    float w = aux[(size_t)m * auxStride + auxOff];
                    float* C = (float*)Cv;
                    C[idx] = C[idx] + w * v;
                }
            }
        }
    }
}

// ---------------------------------------------------------------------------
// RMSNorm (fp32 in, bf16 out). One block per token row.
// ---------------------------------------------------------------------------
__global__ __launch_bounds__(256)
void rmsnorm_bf16_kernel(const float* __restrict__ x, const float* __restrict__ w,
                         bf16* __restrict__ out, int ncols)
{
    __shared__ float red[256];
    int t = blockIdx.x;
    const float* xr = x + (size_t)t * ncols;
    float s = 0.f;
    for (int h = threadIdx.x; h < ncols; h += 256) { float v = xr[h]; s += v * v; }
    red[threadIdx.x] = s;
    __syncthreads();
    for (int off = 128; off > 0; off >>= 1) {
        if (threadIdx.x < off) red[threadIdx.x] += red[threadIdx.x + off];
        __syncthreads();
    }
    float rs = rsqrtf(red[0] / (float)ncols + EPS_);
    for (int h = threadIdx.x; h < ncols; h += 256)
        out[(size_t)t * ncols + h] = f2bf(xr[h] * rs * w[h]);
}

// ---------------------------------------------------------------------------
// RoPE (fp32 in, bf16 out). One block per (token, head), 64 threads.
// ---------------------------------------------------------------------------
__global__ void rope_bf16_kernel(const float* __restrict__ src,
                                 bf16* __restrict__ dst, int nheads)
{
    int bh   = blockIdx.x;
    int tok  = bh / nheads;
    int head = bh % nheads;
    int d    = threadIdx.x;          // 0..63
    int pos  = tok % S_;             // position_ids == arange(S) broadcast
    size_t base = (size_t)tok * nheads * HD_ + (size_t)head * HD_;
    float v = src[base + d];
    int   i = d & 31;
    float freq = powf(THETA_, -(float)(2 * i) / (float)HD_);
    float ang  = (float)pos * freq;
    float c = cosf(ang), s = sinf(ang);
    float other = (d < 32) ? -src[base + d + 32] : src[base + d - 32];
    dst[base + d] = f2bf(v * c + other * s);
}

__global__ void f32_to_bf16_kernel(const float* __restrict__ src,
                                   bf16* __restrict__ dst, size_t n)
{
    size_t i = (size_t)blockIdx.x * blockDim.x + threadIdx.x;
    if (i < n) dst[i] = f2bf(src[i]);
}

__global__ void copy_f32_kernel(const float* __restrict__ src,
                                float* __restrict__ dst, size_t n)
{
    size_t i = ((size_t)blockIdx.x * blockDim.x + threadIdx.x) * 4;
    if (i + 3 < n) *(float4*)(dst + i) = *(const float4*)(src + i);
}

// ---------------------------------------------------------------------------
// Flash-style causal GQA attention with WMMA for QK^T and P*V.
// Block: 128 threads (4 waves), 64 q rows, 32-key chunks, HD=64.
// ---------------------------------------------------------------------------
#define ATT_THREADS 128

__global__ __launch_bounds__(ATT_THREADS)
void attn_wmma_kernel(const bf16* __restrict__ q, const bf16* __restrict__ k,
                      const bf16* __restrict__ v, bf16* __restrict__ out)
{
    __shared__ __align__(32) bf16 Qs[64][64];
    __shared__ __align__(32) bf16 Ks[32][64];   // [key][d] == B-frag of K^T
    __shared__ __align__(32) bf16 Vt[64][32];   // [d][key] == B-frag of V
    __shared__ __align__(32) bf16 Ps[64][32];
    __shared__ float Ss[64][32];
    __shared__ float Os[64][64];
    __shared__ float mrow[64], lrow[64];

    const int nqb = S_ / 64;
    int qb  = blockIdx.x % nqb;
    int h   = (blockIdx.x / nqb) % NH_;
    int b   = blockIdx.x / (nqb * NH_);
    int kvh = h >> 2;                 // GQA: NH/NKV = 4
    int q0  = qb * 64;

    const int tid  = threadIdx.x;
    const int lane = tid & 31;
    const int wave = tid >> 5;        // 0..3, owns rows [wave*16, +16)
    const int lh   = lane & 15;
    const int lg   = lane >> 4;

    for (int idx = tid; idx < 64 * 64; idx += ATT_THREADS) {
        int r = idx >> 6, d = idx & 63;
        Qs[r][d] = q[((size_t)(b * S_ + q0 + r) * NH_ + h) * HD_ + d];
        ((float*)Os)[idx] = 0.f;
    }
    if (tid < 64) { mrow[tid] = -3.0e38f; lrow[tid] = 0.f; }
    __syncthreads();

    int nkc = (q0 + 64) / 32;         // key chunks touching the causal region
    for (int kc = 0; kc < nkc; kc++) {
        int kb0 = kc * 32;
        for (int idx = tid; idx < 32 * 64; idx += ATT_THREADS) {
            int j = idx >> 6, d = idx & 63;
            size_t so = ((size_t)(b * S_ + kb0 + j) * NKV_ + kvh) * HD_ + d;
            Ks[j][d] = k[so];
            Vt[d][j] = v[so];
        }
        __syncthreads();

        // S = Q x K^T (per wave: 16 rows x 32 keys, HD contracted in 2 steps)
        #pragma unroll
        for (int ni = 0; ni < 2; ni++) {
            v8f acc = vzero8();
            #pragma unroll
            for (int ks = 0; ks < 2; ks++) {
                int mm = wave * 16 + lh;
                v8bf lo = *(const v8bf*)&Qs[mm][ks * 32 + lg * 16];
                v8bf hi = *(const v8bf*)&Qs[mm][ks * 32 + lg * 16 + 8];
                v16bf afrag = cat8(lo, hi);
                v16bf bfrag = *(const v16bf*)&Ks[ni * 16 + lh][ks * 32 + lg * 16];
                acc = __builtin_amdgcn_wmma_f32_16x16x32_bf16(
                    false, afrag, false, bfrag, (short)0, acc, false, false);
            }
            #pragma unroll
            for (int i = 0; i < 8; i++)
                Ss[wave * 16 + lg * 8 + i][ni * 16 + lh] = acc[i];
        }
        __syncthreads();

        // Online softmax: one thread per q row.
        if (tid < 64) {
            int r    = tid;
            int jmax = (q0 + r) - kb0;       // causal: key kb0+j valid iff j<=jmax
            float mold = mrow[r];
            float mx   = mold;
            for (int j = 0; j < 32; j++)
                if (j <= jmax) { float sv = Ss[r][j] * 0.125f; if (sv > mx) mx = sv; }
            float f = __expf(mold - mx);
            for (int d = 0; d < 64; d++) Os[r][d] *= f;
            float lsum = lrow[r] * f;
            for (int j = 0; j < 32; j++) {
                float p = (j <= jmax) ? __expf(Ss[r][j] * 0.125f - mx) : 0.f;
                Ps[r][j] = f2bf(p);
                lsum += p;
            }
            mrow[r] = mx;
            lrow[r] = lsum;
        }
        __syncthreads();

        // O += P x V (contract 32 keys; 4 d-tiles of 16)
        {
            int mm = wave * 16 + lh;
            v8bf lo = *(const v8bf*)&Ps[mm][lg * 16];
            v8bf hi = *(const v8bf*)&Ps[mm][lg * 16 + 8];
            v16bf afrag = cat8(lo, hi);
            #pragma unroll
            for (int ni = 0; ni < 4; ni++) {
                v16bf bfrag = *(const v16bf*)&Vt[ni * 16 + lh][lg * 16];
                v8f acc = __builtin_amdgcn_wmma_f32_16x16x32_bf16(
                    false, afrag, false, bfrag, (short)0, vzero8(), false, false);
                #pragma unroll
                for (int i = 0; i < 8; i++)
                    Os[wave * 16 + lg * 8 + i][ni * 16 + lh] += acc[i];
            }
        }
        __syncthreads();
    }

    if (tid < 64) {
        int r = tid;
        float inv = 1.0f / lrow[r];
        size_t base = (size_t)(b * S_ + q0 + r) * (NH_ * HD_) + (size_t)h * HD_;
        for (int d = 0; d < 64; d++) out[base + d] = f2bf(Os[r][d] * inv);
    }
}

// ---------------------------------------------------------------------------
// Router: logits -> softmax -> top-2 -> normalized weights[t][E] (zeros else).
// ---------------------------------------------------------------------------
__global__ __launch_bounds__(256)
void router_kernel(const bf16* __restrict__ xn, const float* __restrict__ rw,
                   float* __restrict__ weights)
{
    __shared__ float red[256];
    __shared__ float lg[E_];
    int t = blockIdx.x;
    float acc[E_];
    for (int e = 0; e < E_; e++) acc[e] = 0.f;
    for (int hh = threadIdx.x; hh < H_; hh += 256) {
        float xv = (float)xn[(size_t)t * H_ + hh];
        #pragma unroll
        for (int e = 0; e < E_; e++) acc[e] += xv * rw[(size_t)hh * E_ + e];
    }
    for (int e = 0; e < E_; e++) {
        red[threadIdx.x] = acc[e];
        __syncthreads();
        for (int off = 128; off > 0; off >>= 1) {
            if (threadIdx.x < off) red[threadIdx.x] += red[threadIdx.x + off];
            __syncthreads();
        }
        if (threadIdx.x == 0) lg[e] = red[0];
        __syncthreads();
    }
    if (threadIdx.x == 0) {
        float mx = lg[0];
        for (int e = 1; e < E_; e++) mx = fmaxf(mx, lg[e]);
        float p[E_], s = 0.f;
        for (int e = 0; e < E_; e++) { p[e] = __expf(lg[e] - mx); s += p[e]; }
        for (int e = 0; e < E_; e++) p[e] /= s;
        int i1 = 0;
        for (int e = 1; e < E_; e++) if (p[e] > p[i1]) i1 = e;
        int i2 = (i1 == 0) ? 1 : 0;
        for (int e = 0; e < E_; e++) if (e != i1 && p[e] > p[i2]) i2 = e;
        float s2 = p[i1] + p[i2];
        for (int e = 0; e < E_; e++) weights[(size_t)t * E_ + e] = 0.f;
        weights[(size_t)t * E_ + i1] = p[i1] / s2;
        weights[(size_t)t * E_ + i2] = p[i2] / s2;
    }
}

// ---------------------------------------------------------------------------
// Host-side orchestration (all on `stream`, graph-capture safe).
// ---------------------------------------------------------------------------
extern "C" void kernel_launch(void* const* d_in, const int* in_sizes, int n_in,
                              void* d_out, int out_size, void* d_ws, size_t ws_size,
                              hipStream_t stream)
{
    (void)in_sizes; (void)n_in; (void)out_size; (void)ws_size;

    const float* x_in   = (const float*)d_in[0];
    // d_in[1] = position_ids (arange, recomputed on device)
    const float* ln1_w  = (const float*)d_in[2];
    const float* ln2_w  = (const float*)d_in[3];
    const float* wq     = (const float*)d_in[4];
    const float* wk     = (const float*)d_in[5];
    const float* wv     = (const float*)d_in[6];
    const float* wo     = (const float*)d_in[7];
    const float* rw     = (const float*)d_in[8];
    const float* w_gate = (const float*)d_in[9];
    const float* w_up   = (const float*)d_in[10];
    const float* w_down = (const float*)d_in[11];
    float* outp = (float*)d_out;

    char* ws = (char*)d_ws;
    size_t off = 0;
    auto alloc = [&](size_t bytes) -> void* {
        void* p = ws + off;
        off = (off + bytes + 255) & ~(size_t)255;
        return p;
    };
    bf16*  xn1   = (bf16*) alloc((size_t)T_ * H_ * 2);
    float* qf    = (float*)alloc((size_t)T_ * NH_  * HD_ * 4);
    float* kf    = (float*)alloc((size_t)T_ * NKV_ * HD_ * 4);
    float* vf    = (float*)alloc((size_t)T_ * NKV_ * HD_ * 4);
    bf16*  qbb   = (bf16*) alloc((size_t)T_ * NH_  * HD_ * 2);
    bf16*  kbb   = (bf16*) alloc((size_t)T_ * NKV_ * HD_ * 2);
    bf16*  vbb   = (bf16*) alloc((size_t)T_ * NKV_ * HD_ * 2);
    bf16*  attnb = (bf16*) alloc((size_t)T_ * NH_  * HD_ * 2);
    float* hid   = (float*)alloc((size_t)T_ * H_ * 4);
    bf16*  xn2   = (bf16*) alloc((size_t)T_ * H_ * 2);
    float* wts   = (float*)alloc((size_t)T_ * E_ * 4);
    float* gbuf  = (float*)alloc((size_t)T_ * I_ * 4);
    bf16*  abuf  = (bf16*) alloc((size_t)T_ * I_ * 2);

    // --- attention half ---
    rmsnorm_bf16_kernel<<<T_, 256, 0, stream>>>(x_in, ln1_w, xn1, H_);
    gemm_bf16_wmma<MODE_STORE><<<dim3((NH_*HD_)/BN,  T_/BM), 256, 0, stream>>>(
        xn1, wq, qf, nullptr, T_, NH_*HD_, H_, 0, 0);
    gemm_bf16_wmma<MODE_STORE><<<dim3((NKV_*HD_)/BN, T_/BM), 256, 0, stream>>>(
        xn1, wk, kf, nullptr, T_, NKV_*HD_, H_, 0, 0);
    gemm_bf16_wmma<MODE_STORE><<<dim3((NKV_*HD_)/BN, T_/BM), 256, 0, stream>>>(
        xn1, wv, vf, nullptr, T_, NKV_*HD_, H_, 0, 0);
    rope_bf16_kernel<<<T_ * NH_,  HD_, 0, stream>>>(qf, qbb, NH_);
    rope_bf16_kernel<<<T_ * NKV_, HD_, 0, stream>>>(kf, kbb, NKV_);
    {
        size_t n = (size_t)T_ * NKV_ * HD_;
        f32_to_bf16_kernel<<<(unsigned)((n + 255) / 256), 256, 0, stream>>>(vf, vbb, n);
    }
    attn_wmma_kernel<<<B_ * NH_ * (S_/64), ATT_THREADS, 0, stream>>>(qbb, kbb, vbb, attnb);
    gemm_bf16_wmma<MODE_RESID><<<dim3(H_/BN, T_/BM), 256, 0, stream>>>(
        attnb, wo, hid, x_in, T_, H_, NH_*HD_, 0, 0);

    // --- MoE half ---
    rmsnorm_bf16_kernel<<<T_, 256, 0, stream>>>(hid, ln2_w, xn2, H_);
    router_kernel<<<T_, 256, 0, stream>>>(xn2, rw, wts);
    copy_f32_kernel<<<(T_ * H_) / 1024, 256, 0, stream>>>(hid, outp, (size_t)T_ * H_);
    for (int e = 0; e < E_; e++) {
        const float* wg = w_gate + (size_t)e * H_ * I_;
        const float* wu = w_up   + (size_t)e * H_ * I_;
        const float* wd = w_down + (size_t)e * I_ * H_;
        gemm_bf16_wmma<MODE_STORE><<<dim3(I_/BN, T_/BM), 256, 0, stream>>>(
            xn2, wg, gbuf, nullptr, T_, I_, H_, 0, 0);
        gemm_bf16_wmma<MODE_SILU><<<dim3(I_/BN, T_/BM), 256, 0, stream>>>(
            xn2, wu, (void*)abuf, gbuf, T_, I_, H_, 0, 0);
        gemm_bf16_wmma<MODE_MOE><<<dim3(H_/BN, T_/BM), 256, 0, stream>>>(
            abuf, wd, (void*)outp, wts, T_, H_, I_, E_, e);
    }
}
